// FN_InstanceLoss_58643483459993
// MI455X (gfx1250) — matrix operand order repaired and checked by reading, not compiled, not run on hardware
//
#include <hip/hip_runtime.h>

#define NROWS 8192
#define BHALF 4096
#define DIM   256
#define PHI_F 0.95f
#define TINV  2.0f   // 1 / TEMPERATURE

typedef __bf16 bf16_t;
typedef __attribute__((ext_vector_type(16))) __bf16 v16bf;
typedef __attribute__((ext_vector_type(8)))  __bf16 v8bf;
typedef __attribute__((ext_vector_type(8)))  float  v8f;

// ---------------------------------------------------------------------------
// Kernel 1: per pair r in [0,4096): normalize both views (f32 exact), emit
// bf16 hi/lo split rows (row r from view0, row r+B from view1) and the exact
// f32 positive logit (shared by rows r and r+B).
// ---------------------------------------------------------------------------
__global__ void __launch_bounds__(256)
normalize_pos_kernel(const float* __restrict__ feat,
                     bf16_t* __restrict__ Xhi,
                     bf16_t* __restrict__ Xlo,
                     float*  __restrict__ pos_logit) {
  const int wave = threadIdx.x >> 5;
  const int lane = threadIdx.x & 31;
  const int r = blockIdx.x * 8 + wave;

  const float* f0 = feat + (size_t)r * (2 * DIM);   // features[r][0][:]
  const float* f1 = f0 + DIM;                       // features[r][1][:]
  const int k0 = lane * 8;

  float v0[8], v1[8];
  float ss0 = 0.f, ss1 = 0.f, dt = 0.f;
#pragma unroll
  for (int e = 0; e < 8; ++e) {
    v0[e] = f0[k0 + e];
    v1[e] = f1[k0 + e];
    ss0 = fmaf(v0[e], v0[e], ss0);
    ss1 = fmaf(v1[e], v1[e], ss1);
    dt  = fmaf(v0[e], v1[e], dt);
  }
#pragma unroll
  for (int m = 16; m >= 1; m >>= 1) {
    ss0 += __shfl_xor(ss0, m, 32);
    ss1 += __shfl_xor(ss1, m, 32);
    dt  += __shfl_xor(dt,  m, 32);
  }
  const float s0 = 1.0f / fmaxf(sqrtf(ss0), 1e-12f);
  const float s1 = 1.0f / fmaxf(sqrtf(ss1), 1e-12f);

  if (lane == 0) {
    float praw = fminf(fmaxf(dt * s0 * s1, -1.0f), 1.0f);
    pos_logit[r]         = praw * TINV;
    pos_logit[r + BHALF] = praw * TINV;
  }

  bf16_t* h0 = Xhi + (size_t)r * DIM + k0;
  bf16_t* l0 = Xlo + (size_t)r * DIM + k0;
  bf16_t* h1 = Xhi + (size_t)(r + BHALF) * DIM + k0;
  bf16_t* l1 = Xlo + (size_t)(r + BHALF) * DIM + k0;
#pragma unroll
  for (int e = 0; e < 8; ++e) {
    float x0 = v0[e] * s0;
    bf16_t x0h = (bf16_t)x0;
    h0[e] = x0h;
    l0[e] = (bf16_t)(x0 - (float)x0h);
    float x1 = v1[e] * s1;
    bf16_t x1h = (bf16_t)x1;
    h1[e] = x1h;
    l1[e] = (bf16_t)(x1 - (float)x1h);
  }
}

// ---------------------------------------------------------------------------
// Kernel 2: fused Gram + masked online logsumexp.
// Block = 16-row tile (A in registers, shared K=256). 8 waves sweep the 8192
// columns in 16-wide tiles. Split-bf16 WMMA: hi*hi + hi*lo + lo*hi.
// ---------------------------------------------------------------------------
__global__ void __launch_bounds__(256)
gram_lse_kernel(const bf16_t* __restrict__ Xhi,
                const bf16_t* __restrict__ Xlo,
                const float*  __restrict__ pos_logit,
                float* __restrict__ partials) {
  __shared__ float red_mx[8][16];
  __shared__ float red_sm[8][16];

  const int tid  = threadIdx.x;
  const int wave = tid >> 5;
  const int lane = tid & 31;
  const int lm   = lane & 15;      // A: row slot / B: col slot / C: col slot
  const int h    = lane >> 4;      // lane half -> K sub-slice (A/B), row+8 (C)
  const int m_base = blockIdx.x * 16;
  const int m = m_base + lm;       // A row held by this lane

  // ---- load A (16 rows x 256 K, hi+lo) into registers ----
  v16bf ahi[8], alo[8];
  {
    const bf16_t* arh = Xhi + (size_t)m * DIM;
    const bf16_t* arl = Xlo + (size_t)m * DIM;
#pragma unroll
    for (int c = 0; c < 8; ++c) {
      const int kc = c * 32;
      v8bf p0 = *(const v8bf*)(arh + kc + 8 * h);
      v8bf p1 = *(const v8bf*)(arh + kc + 16 + 8 * h);
      ahi[c] = __builtin_shufflevector(p0, p1, 0, 1, 2, 3, 4, 5, 6, 7,
                                       8, 9, 10, 11, 12, 13, 14, 15);
      v8bf q0 = *(const v8bf*)(arl + kc + 8 * h);
      v8bf q1 = *(const v8bf*)(arl + kc + 16 + 8 * h);
      alo[c] = __builtin_shufflevector(q0, q1, 0, 1, 2, 3, 4, 5, 6, 7,
                                       8, 9, 10, 11, 12, 13, 14, 15);
    }
  }

  float mx[8], sm[8];
#pragma unroll
  for (int r = 0; r < 8; ++r) { mx[r] = -__builtin_inff(); sm[r] = 0.0f; }

  for (int t = 0; t < (NROWS / 16) / 8; ++t) {   // 64 column tiles per wave
    const int n_base = (wave + 8 * t) * 16;
    const int j = n_base + lm;                   // B column == row j of X
    const bf16_t* brh = Xhi + (size_t)j * DIM + 8 * h;
    const bf16_t* brl = Xlo + (size_t)j * DIM + 8 * h;

    v8f acc = {};
#pragma unroll
    for (int c = 0; c < 8; ++c) {
      const int kc = c * 32;
      v8bf b0 = *(const v8bf*)(brh + kc);
      v8bf b1 = *(const v8bf*)(brh + kc + 16);
      v16bf bhi = __builtin_shufflevector(b0, b1, 0, 1, 2, 3, 4, 5, 6, 7,
                                          8, 9, 10, 11, 12, 13, 14, 15);
      v8bf b2 = *(const v8bf*)(brl + kc);
      v8bf b3 = *(const v8bf*)(brl + kc + 16);
      v16bf blo = __builtin_shufflevector(b2, b3, 0, 1, 2, 3, 4, 5, 6, 7,
                                          8, 9, 10, 11, 12, 13, 14, 15);
      acc = __builtin_amdgcn_wmma_f32_16x16x32_bf16(false, ahi[c], false, bhi,
                                                    (short)0, acc, false, false);
      acc = __builtin_amdgcn_wmma_f32_16x16x32_bf16(false, ahi[c], false, blo,
                                                    (short)0, acc, false, false);
      acc = __builtin_amdgcn_wmma_f32_16x16x32_bf16(false, alo[c], false, bhi,
                                                    (short)0, acc, false, false);
    }

#pragma unroll
    for (int r = 0; r < 8; ++r) {
      const int mr = m_base + r + 8 * h;         // C row for this lane/reg
      float raw   = fminf(fmaxf(acc[r], -1.0f), 1.0f);
      float logit = raw * TINV;
      bool keep = (j != mr) && (j != (mr ^ BHALF)) && (raw < PHI_F);
      if (keep) {
        if (logit <= mx[r]) {
          sm[r] += __expf(logit - mx[r]);
        } else {
          sm[r] = sm[r] * __expf(mx[r] - logit) + 1.0f;
          mx[r] = logit;
        }
      }
    }
  }

  // ---- combine across the 16 lanes sharing each row half ----
#pragma unroll
  for (int mk = 1; mk <= 8; mk <<= 1) {
#pragma unroll
    for (int r = 0; r < 8; ++r) {
      float omx = __shfl_xor(mx[r], mk, 32);
      float osm = __shfl_xor(sm[r], mk, 32);
      float nm  = fmaxf(mx[r], omx);
      sm[r] = sm[r] * __expf(mx[r] - nm) + osm * __expf(omx - nm);
      mx[r] = nm;
    }
  }
  if (lm == 0) {
#pragma unroll
    for (int r = 0; r < 8; ++r) {
      red_mx[wave][r + 8 * h] = mx[r];
      red_sm[wave][r + 8 * h] = sm[r];
    }
  }
  __syncthreads();

  // ---- cross-wave merge + positive term + per-block partial ----
  if (tid < 16) {
    float cmx = -__builtin_inff(), csm = 0.0f;
#pragma unroll
    for (int w = 0; w < 8; ++w) {
      float omx = red_mx[w][tid], osm = red_sm[w][tid];
      float nm  = fmaxf(cmx, omx);
      csm = csm * __expf(cmx - nm) + osm * __expf(omx - nm);
      cmx = nm;
    }
    float pl  = pos_logit[m_base + tid];
    float nm  = fmaxf(cmx, pl);
    float den = nm + __logf(csm * __expf(cmx - nm) + __expf(pl - nm));
    float contrib = den - pl;
#pragma unroll
    for (int mk = 8; mk >= 1; mk >>= 1)
      contrib += __shfl_xor(contrib, mk, 32);
    if (tid == 0) partials[blockIdx.x] = contrib;
  }
}

// ---------------------------------------------------------------------------
// Kernel 3: sum 512 partials -> mean
// ---------------------------------------------------------------------------
__global__ void __launch_bounds__(256)
finalize_kernel(const float* __restrict__ partials, float* __restrict__ out) {
  __shared__ float wsum[8];
  const int tid = threadIdx.x;
  float v = partials[tid] + partials[tid + 256];
#pragma unroll
  for (int mk = 16; mk >= 1; mk >>= 1) v += __shfl_xor(v, mk, 32);
  if ((tid & 31) == 0) wsum[tid >> 5] = v;
  __syncthreads();
  if (tid == 0) {
    float s = 0.0f;
#pragma unroll
    for (int w = 0; w < 8; ++w) s += wsum[w];
    out[0] = s / (float)NROWS;
  }
}

// ---------------------------------------------------------------------------
extern "C" void kernel_launch(void* const* d_in, const int* in_sizes, int n_in,
                              void* d_out, int out_size, void* d_ws, size_t ws_size,
                              hipStream_t stream) {
  (void)in_sizes; (void)n_in; (void)out_size; (void)ws_size;
  const float* feat = (const float*)d_in[0];
  float* out = (float*)d_out;

  char* ws = (char*)d_ws;
  const size_t xbytes = (size_t)NROWS * DIM * sizeof(bf16_t);  // 4 MB each
  bf16_t* Xhi = (bf16_t*)ws;
  bf16_t* Xlo = (bf16_t*)(ws + xbytes);
  float*  pos = (float*)(ws + 2 * xbytes);
  float*  partials = pos + NROWS;

  normalize_pos_kernel<<<BHALF / 8, 256, 0, stream>>>(feat, Xhi, Xlo, pos);
  gram_lse_kernel<<<NROWS / 16, 256, 0, stream>>>(Xhi, Xlo, pos, partials);
  finalize_kernel<<<1, 256, 0, stream>>>(partials, out);
}
